// ResBlock_53747220742201
// MI455X (gfx1250) — matrix operand order repaired and checked by reading, not compiled, not run on hardware
//
#include <hip/hip_runtime.h>
#include <math.h>

// ---------------- CDNA5 WMMA types/helpers (gfx1250, wave32) ----------------
typedef __attribute__((ext_vector_type(16))) __bf16 v16bf;
typedef __attribute__((ext_vector_type(8)))  float  v8f;
typedef unsigned int __attribute__((ext_vector_type(4))) u32x4;
typedef int          __attribute__((ext_vector_type(8))) i32x8;
typedef int          __attribute__((ext_vector_type(4))) i32x4;

#define B_    4
#define C_    128
#define HH    64
#define WW    64
#define HWSZ  4096
#define GROUPS_ 32
#define TDIM  512

#ifndef __has_builtin
#define __has_builtin(x) 0
#endif
#if __has_builtin(__builtin_amdgcn_tensor_load_to_lds) && __has_builtin(__builtin_amdgcn_s_wait_tensorcnt)
#define HAVE_TDM 1
#else
#define HAVE_TDM 0
#endif

__device__ __forceinline__ float silu(float v){ return v / (1.0f + __expf(-v)); }

#if HAVE_TDM
// Generic pointers into the LDS aperture carry the LDS byte offset in their low
// 32 bits (ISA 10.2).
__device__ __forceinline__ unsigned lds_off(const void* p){
  return (unsigned)(unsigned long long)p;
}
// Issue a 2D TDM tile load: tile (tile0 x tile1) of bf16 (data_size=2B) from a
// row-major tensor with row stride stride0 (elements) into LDS at lds_byte_off.
// d0g1 carries {data_size, pad_enable, pad_interval, pad_amount} per D# group1.
// This toolchain (clang-23 / therock-10.0) exposes the 6-arg builtin:
//   (uint32x4 g0, int32x8 g1, int32x4 g2, int32x4 g3, int32x8 pad, i32 cpol)
__device__ __forceinline__ void tdm_load_2d(unsigned lds_byte_off, unsigned long long gaddr,
                                            unsigned d0g1, unsigned tile0, unsigned tile1,
                                            unsigned long long stride0){
  const unsigned BIG = 1u << 20;          // tensor dims >> tile: no OOB clipping
  u32x4 g0;
  g0[0] = 1u;                              // count=1, user descriptor, no gather
  g0[1] = lds_byte_off;                    // D#.lds_addr
  g0[2] = (unsigned)(gaddr & 0xFFFFFFFFull);             // global_addr[31:0]
  g0[3] = (unsigned)((gaddr >> 32) & 0x1FFFFFFull) | (2u << 30); // [56:32] | type=2
  i32x8 g1;
  g1[0] = (int)d0g1;                                      // wg_mask=0, data_size, pad cfg
  g1[1] = (int)((BIG & 0xFFFFu) << 16);                   // tensor_dim0[15:0]
  g1[2] = (int)(((BIG >> 16) & 0xFFFFu) | ((BIG & 0xFFFFu) << 16));  // td0 hi | td1 lo
  g1[3] = (int)(((BIG >> 16) & 0xFFFFu) | (tile0 << 16)); // td1 hi | tile_dim0
  g1[4] = (int)(tile1 & 0xFFFFu);                         // tile_dim1, tile_dim2=0
  g1[5] = (int)(unsigned)(stride0 & 0xFFFFFFFFull);       // tensor_dim0_stride lo
  g1[6] = (int)(unsigned)((stride0 >> 32) & 0xFFFFull);   // stride0 hi | stride1 lo=0
  g1[7] = 0;
  i32x4 z4 = {0, 0, 0, 0};
  i32x8 z8 = {0, 0, 0, 0, 0, 0, 0, 0};
  __builtin_amdgcn_tensor_load_to_lds(g0, g1, z4, z4, z8, 0);
}
// pad cfg: row = tile0*2 bytes, pad_interval chosen so hardware inserts
// pad_amount DWORDs after each row -> padded LDS leading dimension.
#define TDM_D0_ROW64B_PAD1DW  ((1u<<16) | (1u<<20) | (3u<<22) | (0u<<25))  // ld 32+2
#define TDM_D0_ROW128B_PAD2DW ((1u<<16) | (1u<<20) | (4u<<22) | (1u<<25))  // ld 64+4
#define TDM_D0_ROW256B_PAD2DW ((1u<<16) | (1u<<20) | (5u<<22) | (1u<<25))  // ld 128+4
#endif

// A fragment: A stored row-major [M][lda] bf16 (contiguous k order: lane half h holds k = 16h + i).
__device__ __forceinline__ v16bf frag_a(const __bf16* A, int lda, int m0, int k0, int lane){
  int m  = m0 + (lane & 15);
  int kh = (lane >> 4) * 16;
  const __bf16* p = A + m * lda + k0 + kh;
  v16bf a;
#pragma unroll
  for (int i = 0; i < 16; ++i) a[i] = p[i];
  return a;
}

// B fragment, B stored [N][ldb] (K contiguous per row).
__device__ __forceinline__ v16bf frag_b_kcontig(const __bf16* Bm, int ldb, int n0, int k0, int lane){
  int n  = n0 + (lane & 15);
  int kh = (lane >> 4) * 16;
  const __bf16* p = Bm + n * ldb + k0 + kh;
  v16bf b;
#pragma unroll
  for (int i = 0; i < 16; ++i) b[i] = p[i];
  return b;
}

// B fragment, B stored [K][ldb] (N contiguous, K strided).
__device__ __forceinline__ v16bf frag_b_kstrided(const __bf16* Bm, int ldb, int n0, int k0, int lane){
  int n  = n0 + (lane & 15);
  int kh = (lane >> 4) * 16;
  const __bf16* p = Bm + (k0 + kh) * ldb + n;
  v16bf b;
#pragma unroll
  for (int i = 0; i < 16; ++i) b[i] = p[i * ldb];
  return b;
}

__device__ __forceinline__ v8f wmma_bf16(v16bf a, v16bf b, v8f c){
  return __builtin_amdgcn_wmma_f32_16x16x32_bf16(false, a, false, b, (short)0, c, false, false);
}

// ---------------- GroupNorm(32) + SiLU, one block per (b, group) ----------------
__global__ void gn_silu_kernel(const float* __restrict__ x, const float* __restrict__ scale,
                               const float* __restrict__ bias, float* __restrict__ out){
  int b = blockIdx.x >> 5;
  int g = blockIdx.x & 31;
  const int CPG = C_ / GROUPS_;
  const int N   = CPG * HWSZ;
  size_t base = (size_t)(b * C_ + g * CPG) * HWSZ;
  float s = 0.f, sq = 0.f;
  for (int i = threadIdx.x; i < N; i += 256){
    float v = x[base + i];
    s += v; sq += v * v;
  }
  __shared__ float rs[256], rq[256];
  rs[threadIdx.x] = s; rq[threadIdx.x] = sq;
  __syncthreads();
  for (int off = 128; off > 0; off >>= 1){
    if (threadIdx.x < off){
      rs[threadIdx.x] += rs[threadIdx.x + off];
      rq[threadIdx.x] += rq[threadIdx.x + off];
    }
    __syncthreads();
  }
  float mean = rs[0] / (float)N;
  float var  = rq[0] / (float)N - mean * mean;
  float rstd = rsqrtf(var + 1e-5f);
  for (int i = threadIdx.x; i < N; i += 256){
    int c = g * CPG + (i >> 12);
    float v = (x[base + i] - mean) * rstd * scale[c] + bias[c];
    out[base + i] = silu(v);
  }
}

// ---------------- temb = SiLU(t) @ mlp_w^T + mlp_b ----------------
__global__ void temb_kernel(const float* __restrict__ t, const float* __restrict__ w,
                            const float* __restrict__ bias, float* __restrict__ temb){
  int idx = blockIdx.x * 256 + threadIdx.x;
  if (idx >= B_ * C_) return;
  int b = idx / C_, co = idx % C_;
  float s = 0.f;
  for (int k = 0; k < TDIM; ++k) s += silu(t[b * TDIM + k]) * w[co * TDIM + k];
  temb[idx] = s + bias[co];
}

// ---------------- pack conv weights f32[co][ci][3][3] -> bf16[tap][co][ci] ----------------
__global__ void pack_w_kernel(const float* __restrict__ w, __bf16* __restrict__ wpk){
  int idx = blockIdx.x * 256 + threadIdx.x;
  if (idx >= C_ * C_ * 9) return;
  int ci  = idx % C_;
  int co  = (idx / C_) % C_;
  int tap = idx / (C_ * C_);
  wpk[idx] = (__bf16)w[(co * C_ + ci) * 9 + tap];
}

// ---------------- 1x1 conv (cond projection) -> bf16 output for TDM/WMMA ----------------
__global__ void cond1x1_kernel(const float* __restrict__ cond, const float* __restrict__ w,
                               const float* __restrict__ bias, __bf16* __restrict__ outbf){
  int b = blockIdx.x >> 6;
  int h = blockIdx.x & 63;
  __shared__ float tile[C_][WW];
  for (int idx = threadIdx.x; idx < C_ * WW; idx += 256){
    int ww = idx & 63, ci = idx >> 6;
    tile[ci][ww] = cond[((size_t)(b * C_ + ci) * HH + h) * WW + ww];
  }
  __syncthreads();
  int ww = threadIdx.x & 63;
  int cb = threadIdx.x >> 6;
  for (int co = cb * 32; co < cb * 32 + 32; ++co){
    float s = 0.f;
    for (int ci = 0; ci < C_; ++ci) s += w[co * C_ + ci] * tile[ci][ww];
    outbf[((size_t)(b * C_ + co) * HH + h) * WW + ww] = (__bf16)(s + bias[co]);
  }
}

// ---------------- 3x3 conv as implicit GEMM with WMMA bf16 ----------------
// One block per (b, output row h). Per tap: weight tile 128x128 arrives via TDM
// (hardware-padded to ld=132); im2col row tile staged by threads (needs f32->bf16).
__global__ void conv3x3_kernel(const float* __restrict__ in, const __bf16* __restrict__ wpk,
                               const float* __restrict__ bias, const float* __restrict__ addc,
                               const float* __restrict__ addres, float* __restrict__ out){
  int b = blockIdx.x >> 6;
  int h = blockIdx.x & 63;
  int tid = threadIdx.x, lane = tid & 31, wave = tid >> 5;
  __shared__ __bf16 Asm[C_][132];        // weights [co][ci], TDM-padded ld
  __shared__ __bf16 Bsm[WW][132];        // im2col  [pixel][ci]
  v8f acc[4] = {};
  for (int tap = 0; tap < 9; ++tap){
    int dh = tap / 3 - 1, dw = tap % 3 - 1;
    int hh = h + dh;
    bool hok = (hh >= 0) && (hh < HH);
#if HAVE_TDM
    if (wave == 0)
      tdm_load_2d(lds_off(&Asm[0][0]),
                  (unsigned long long)(size_t)(wpk + (size_t)tap * C_ * C_),
                  TDM_D0_ROW256B_PAD2DW, C_, C_, C_);
#else
    for (int idx = tid; idx < C_ * C_; idx += 256){
      int ci = idx & 127, co = idx >> 7;
      Asm[co][ci] = wpk[(size_t)tap * C_ * C_ + idx];
    }
#endif
    for (int idx = tid; idx < WW * C_; idx += 256){
      int n = idx & 63, ci = idx >> 6;
      int ww = n + dw;
      float v = 0.f;
      if (hok && ww >= 0 && ww < WW)
        v = in[((size_t)(b * C_ + ci) * HH + hh) * WW + ww];
      Bsm[n][ci] = (__bf16)v;
    }
#if HAVE_TDM
    if (wave == 0) __builtin_amdgcn_s_wait_tensorcnt(0);
#endif
    __syncthreads();
#pragma unroll
    for (int k0 = 0; k0 < C_; k0 += 32){
      v16bf af = frag_a(&Asm[0][0], 132, wave * 16, k0, lane);
#pragma unroll
      for (int nt = 0; nt < 4; ++nt){
        v16bf bf = frag_b_kcontig(&Bsm[0][0], 132, nt * 16, k0, lane);
        acc[nt] = wmma_bf16(af, bf, acc[nt]);
      }
    }
    __syncthreads();
  }
  int co0 = wave * 16;
#pragma unroll
  for (int nt = 0; nt < 4; ++nt){
    int n = nt * 16 + (lane & 15);
#pragma unroll
    for (int r = 0; r < 8; ++r){
      int co = co0 + r + (lane >> 4) * 8;
      size_t o = ((size_t)(b * C_ + co) * HH + h) * WW + n;
      float v = acc[nt][r] + bias[co];
      if (addc)   v += addc[b * C_ + co];
      if (addres) v += addres[o];
      out[o] = v;
    }
  }
}

// ---------------- fused spatial attention: one block per (b, w) ----------------
// GEMM1: S[h,uv] = A(64x128) x C(128x64); exact column softmax over h (fully
// resident, axis length 64); GEMM2 computed transposed, Out^T[co,h] += C x P^T,
// so both GEMM2 fragments are k-contiguous against the TDM-loaded Craw[c][uv].
__global__ void attn_kernel(const float* __restrict__ h1, const __bf16* __restrict__ cmbf,
                            const float* __restrict__ ce, const int* __restrict__ aim,
                            float* __restrict__ h2){
  int b = blockIdx.x >> 6;
  int w = blockIdx.x & 63;
  int tid = threadIdx.x, lane = tid & 31, wave = tid >> 5;
  __shared__ __bf16 Asm[HH][132];        // [h][c]   = h1 slice at fixed w
  __shared__ __bf16 Craw[C_][68];        // [c][uv]  TDM tile, hw-padded ld=68
  __shared__ float  Ssm[HH][65];         // scores f32
  __shared__ __bf16 Psm[HH][68];         // probs [h][uv]
  for (int idx = tid; idx < HH * C_; idx += 256){
    int hh = idx & 63, c = idx >> 6;
    Asm[hh][c] = (__bf16)h1[(size_t)(b * C_ + c) * HWSZ + hh * WW + w];
  }
  v8f oacc[4] = {};
  const float rscale = 0.08838834764831845f;   // 1/sqrt(128)
  for (int uv0 = 0; uv0 < HWSZ; uv0 += 64){
#if HAVE_TDM
    if (wave == 0){
      tdm_load_2d(lds_off(&Craw[0][0]),
                  (unsigned long long)(size_t)(cmbf + (size_t)b * C_ * HWSZ + uv0),
                  TDM_D0_ROW128B_PAD2DW, 64, C_, HWSZ);
      __builtin_amdgcn_s_wait_tensorcnt(0);
    }
#else
    for (int idx = tid; idx < 64 * C_; idx += 256){
      int n = idx & 63, c = idx >> 6;
      Craw[c][n] = cmbf[(size_t)(b * C_ + c) * HWSZ + uv0 + n];
    }
#endif
    __syncthreads();
    // GEMM1: 16 tiles of S(64x64), 2 per wave; B read k-strided from Craw.
#pragma unroll
    for (int t = 0; t < 2; ++t){
      int id = wave * 2 + t;
      int mi = id >> 2, ni = id & 3;
      v8f sacc = {};
#pragma unroll
      for (int k0 = 0; k0 < C_; k0 += 32){
        v16bf af = frag_a(&Asm[0][0], 132, mi * 16, k0, lane);
        v16bf bf = frag_b_kstrided(&Craw[0][0], 68, ni * 16, k0, lane);
        sacc = wmma_bf16(af, bf, sacc);
      }
      int n = ni * 16 + (lane & 15);
#pragma unroll
      for (int r = 0; r < 8; ++r){
        int m = mi * 16 + r + (lane >> 4) * 8;
        Ssm[m][n] = sacc[r] * rscale;
      }
    }
    __syncthreads();
    // exact softmax over h per uv column
    if (tid < 64){
      int col = tid;
      float mx = -1e30f;
      for (int hh = 0; hh < HH; ++hh) mx = fmaxf(mx, Ssm[hh][col]);
      float sum = 0.f;
      for (int hh = 0; hh < HH; ++hh){
        float e = __expf(Ssm[hh][col] - mx);
        Ssm[hh][col] = e; sum += e;
      }
      float inv = 1.f / sum;
      for (int hh = 0; hh < HH; ++hh) Psm[hh][col] = (__bf16)(Ssm[hh][col] * inv);
    }
    __syncthreads();
    // GEMM2 (transposed): Out^T(128co x 64h) += Craw(128x64uv) x P^T(64uv x 64h)
    // A k-contig from Craw rows, B k-contig from Psm rows. Wave owns co-tile.
#pragma unroll
    for (int k0 = 0; k0 < 64; k0 += 32){
      v16bf af = frag_a(&Craw[0][0], 68, wave * 16, k0, lane);
#pragma unroll
      for (int nt = 0; nt < 4; ++nt){
        v16bf bf = frag_b_kcontig(&Psm[0][0], 68, nt * 16, k0, lane);
        oacc[nt] = wmma_bf16(af, bf, oacc[nt]);
      }
    }
    __syncthreads();
  }
  const float* ceb = ce + aim[b] * C_;
#pragma unroll
  for (int nt = 0; nt < 4; ++nt){
    int hh = nt * 16 + (lane & 15);
#pragma unroll
    for (int r = 0; r < 8; ++r){
      int co = wave * 16 + r + (lane >> 4) * 8;
      size_t o = (size_t)(b * C_ + co) * HWSZ + hh * WW + w;
      h2[o] = h1[o] + oacc[nt][r] + ceb[co];
    }
  }
}

// ---------------- host orchestration ----------------
extern "C" void kernel_launch(void* const* d_in, const int* in_sizes, int n_in,
                              void* d_out, int out_size, void* d_ws, size_t ws_size,
                              hipStream_t stream){
  const float* x       = (const float*)d_in[0];
  const float* t       = (const float*)d_in[1];
  const int*   aim     = (const int*)d_in[2];
  const float* cond    = (const float*)d_in[3];
  const float* gn1_s   = (const float*)d_in[4];
  const float* gn1_b   = (const float*)d_in[5];
  const float* conv1_w = (const float*)d_in[6];
  const float* conv1_b = (const float*)d_in[7];
  const float* mlp_w   = (const float*)d_in[8];
  const float* mlp_b   = (const float*)d_in[9];
  const float* cond_w  = (const float*)d_in[10];
  const float* cond_b  = (const float*)d_in[11];
  const float* gn2_s   = (const float*)d_in[12];
  const float* gn2_b   = (const float*)d_in[13];
  const float* conv2_w = (const float*)d_in[14];
  const float* conv2_b = (const float*)d_in[15];
  const float* ce      = (const float*)d_in[16];
  float* out = (float*)d_out;

  char* ws = (char*)d_ws;
  const size_t TEN = (size_t)B_ * C_ * HWSZ * sizeof(float);   // 8 MB
  float*  a1   = (float*)(ws + 0 * TEN);   // GN1+SiLU(x)
  float*  h1   = (float*)(ws + 1 * TEN);   // conv1 + temb
  float*  h2   = (float*)(ws + 2 * TEN);   // + attention + contrast
  float*  a2   = (float*)(ws + 3 * TEN);   // GN2+SiLU(h2)
  float*  temb = (float*)(ws + 4 * TEN);
  __bf16* w1   = (__bf16*)(ws + 4 * TEN + 4096);
  __bf16* w2   = w1 + C_ * C_ * 9;
  __bf16* cmbf = w2 + C_ * C_ * 9;         // 1x1 projection, bf16 (4 MB)

  pack_w_kernel<<<576, 256, 0, stream>>>(conv1_w, w1);
  pack_w_kernel<<<576, 256, 0, stream>>>(conv2_w, w2);
  gn_silu_kernel<<<B_ * GROUPS_, 256, 0, stream>>>(x, gn1_s, gn1_b, a1);
  temb_kernel<<<2, 256, 0, stream>>>(t, mlp_w, mlp_b, temb);
  cond1x1_kernel<<<B_ * HH, 256, 0, stream>>>(cond, cond_w, cond_b, cmbf);
  conv3x3_kernel<<<B_ * HH, 256, 0, stream>>>(a1, w1, conv1_b, temb, nullptr, h1);
  attn_kernel<<<B_ * WW, 256, 0, stream>>>(h1, cmbf, ce, aim, h2);
  gn_silu_kernel<<<B_ * GROUPS_, 256, 0, stream>>>(h2, gn2_s, gn2_b, a2);
  conv3x3_kernel<<<B_ * HH, 256, 0, stream>>>(a2, w2, conv2_b, nullptr, x, out);
}
